// RBFD_28089086116331
// MI455X (gfx1250) — matrix operand order repaired and checked by reading, not compiled, not run on hardware
//
#include <hip/hip_runtime.h>

// ---------------------------------------------------------------------------
// Problem constants
// ---------------------------------------------------------------------------
constexpr int N = 320;
constexpr int D = 64;

typedef __attribute__((ext_vector_type(16))) _Float16 v16h;
typedef __attribute__((ext_vector_type(8)))  float    v8f;

union FragU { v16h v; uint4 q[2]; };

// ---------------------------------------------------------------------------
// Kernel A: all N^2-sized quantities. One wave per (i,j) pair, each lane owns
// d = lane and d = lane+32, butterfly reduction across the wave32.
// Writes: k, k_xj, k_xi, tr_k_xj_xi, v_t_k_xj_xi_grad_v_score
// Workspace: t2_1[i,j] = sum_d gamma^3 od^2,  C[i,j] = -2 k sum_d score*gamma*god
// Streaming outputs (never re-read) use non-temporal stores; k/t2/C are
// re-read by kernel B and stay regular-temporal.
// ---------------------------------------------------------------------------
__global__ __launch_bounds__(256) void pairwise_kernel(
    const float* __restrict__ x, const float* __restrict__ sigma,
    const float* __restrict__ score, const float* __restrict__ v,
    const float* __restrict__ gvs,
    float* __restrict__ out_k, float* __restrict__ out_kxj,
    float* __restrict__ out_kxi, float* __restrict__ out_tr,
    float* __restrict__ out_v,
    float* __restrict__ ws_t2, float* __restrict__ ws_C)
{
    const int lane = threadIdx.x & 31;
    const int wid  = threadIdx.x >> 5;
    const int i = blockIdx.y;
    const int j = blockIdx.x * 8 + wid;

    float sk = 0.f, sg2 = 0.f, st2 = 0.f, sa = 0.f, sb = 0.f,
          sc = 0.f, ssc = 0.f, sgs = 0.f;
    float god0, god1;

    // d = lane
    {
        const int d = lane;
        float s  = sigma[d];
        float g  = 1.0f / (2.0f * s * s + 1e-8f);
        float od = x[i * D + d] - x[j * D + d];
        float gd = g * od;                       // god
        god0 = gd;
        float gd2 = gd * gd;                     // gods = gamma^2 od^2
        sk  += g * od * od;                      // gamma * od^2
        sg2 += gd2;
        st2 += g * gd2;                          // gamma * gods
        float vi = v[i * D + d], gv = gvs[i * D + d];
        sa  += vi * gd;
        sb  += gv * gd;
        sc  += g * vi * gv;
        ssc += score[j * D + d] * g * gd;        // score * gamma * god
        sgs += g;
    }
    // d = lane + 32
    {
        const int d = lane + 32;
        float s  = sigma[d];
        float g  = 1.0f / (2.0f * s * s + 1e-8f);
        float od = x[i * D + d] - x[j * D + d];
        float gd = g * od;
        god1 = gd;
        float gd2 = gd * gd;
        sk  += g * od * od;
        sg2 += gd2;
        st2 += g * gd2;
        float vi = v[i * D + d], gv = gvs[i * D + d];
        sa  += vi * gd;
        sb  += gv * gd;
        sc  += g * vi * gv;
        ssc += score[j * D + d] * g * gd;
        sgs += g;
    }

    // wave32 butterfly reduction -> every lane holds the full sums
    #pragma unroll
    for (int off = 16; off >= 1; off >>= 1) {
        sk  += __shfl_xor(sk,  off, 32);
        sg2 += __shfl_xor(sg2, off, 32);
        st2 += __shfl_xor(st2, off, 32);
        sa  += __shfl_xor(sa,  off, 32);
        sb  += __shfl_xor(sb,  off, 32);
        sc  += __shfl_xor(sc,  off, 32);
        ssc += __shfl_xor(ssc, off, 32);
        sgs += __shfl_xor(sgs, off, 32);
    }

    const float kk = expf(-sk);
    const size_t pij = (size_t)i * N + j;

    if (lane == 0) {
        out_k[pij]  = kk;                                  // re-read by kernel B
        ws_t2[pij]  = st2;                                 // re-read by kernel B
        ws_C[pij]   = -2.0f * kk * ssc;                    // re-read by kernel B
        __builtin_nontemporal_store(2.0f * kk * (sgs - 2.0f * sg2), &out_tr[pij]);
        __builtin_nontemporal_store(2.0f * kk * (sc - 2.0f * sa * sb), &out_v[pij]);
    }

    const float t0 = 2.0f * kk * god0;
    const float t1 = 2.0f * kk * god1;
    __builtin_nontemporal_store( t0, &out_kxj[pij * D + lane]);
    __builtin_nontemporal_store( t1, &out_kxj[pij * D + lane + 32]);
    __builtin_nontemporal_store(-t0, &out_kxi[pij * D + lane]);
    __builtin_nontemporal_store(-t1, &out_kxi[pij * D + lane + 32]);
}

// ---------------------------------------------------------------------------
// Kernel B: the two [N,N,N] outputs, fused. Per block: batch i, a 32x32 (r,c)
// tile. Two shared-B GEMMs over K=D=64 via v_wmma_f32_16x16x32_f16:
//   M[r,c] = god_r . god_c   (f16 in, f32 accumulate)
//   A[r,c] = score_r . god_c
// Epilogue:
//   out5[i,r,c] = 2*(C[i,r] + 4*k[i,r]*A*M) * k[i,c]
//   out6[i,r,c] = 4*(4*M^2 - 2*(t2[i,r]+t2[i,c]) + sum(gamma^2)) * k[i,r]*k[i,c]
// Both outputs are pure streaming writes (262 MB total, > L2) -> NT stores.
// ---------------------------------------------------------------------------
#define LDH 72   // row pitch in halfs: 144B rows keep 16B alignment, spread banks

__global__ __launch_bounds__(128) void nnn_kernel(
    const float* __restrict__ x, const float* __restrict__ sigma,
    const float* __restrict__ score,
    const float* __restrict__ kmat, const float* __restrict__ t2,
    const float* __restrict__ Cws,
    float* __restrict__ out5, float* __restrict__ out6)
{
    __shared__ __align__(16) _Float16 sGr[32][LDH];   // god rows (r range)
    __shared__ __align__(16) _Float16 sGc[32][LDH];   // god rows (c range)
    __shared__ __align__(16) _Float16 sSc[32][LDH];   // score rows (r range)
    __shared__ float sGamma[D];
    __shared__ float sKr[32], sT2r[32], sCr[32], sKc[32], sT2c[32];
    __shared__ float sG2;

    const int tid = threadIdx.x;
    const int i   = blockIdx.z;
    const int r0  = blockIdx.y * 32;
    const int c0  = blockIdx.x * 32;

    if (tid < D) {
        float s = sigma[tid];
        sGamma[tid] = 1.0f / (2.0f * s * s + 1e-8f);
    }
    __syncthreads();

    if (tid == 0) {
        float acc = 0.f;
        for (int d = 0; d < D; ++d) { float g = sGamma[d]; acc += g * g; }
        sG2 = acc;
    }
    if (tid < 32) {
        const size_t br = (size_t)i * N + r0 + tid;
        const size_t bc = (size_t)i * N + c0 + tid;
        sKr[tid]  = kmat[br];  sT2r[tid] = t2[br];  sCr[tid] = Cws[br];
        sKc[tid]  = kmat[bc];  sT2c[tid] = t2[bc];
    }
    {   // stage tiles: 128 threads x 16 elements = 32 rows x 64 cols
        const int row   = tid >> 2;
        const int cbase = (tid & 3) * 16;
        const float* xi = x + (size_t)i * D;
        const float* xr = x + (size_t)(r0 + row) * D;
        const float* xc = x + (size_t)(c0 + row) * D;
        const float* sr = score + (size_t)(r0 + row) * D;
        #pragma unroll
        for (int e = 0; e < 16; ++e) {
            const int d = cbase + e;
            const float g   = sGamma[d];
            const float xid = xi[d];
            sGr[row][d] = (_Float16)(g * (xid - xr[d]));
            sGc[row][d] = (_Float16)(g * (xid - xc[d]));
            sSc[row][d] = (_Float16)(sr[d]);
        }
    }
    __syncthreads();

    const int lane = tid & 31;
    const int wave = tid >> 5;                // 0..3
    const int wr   = (wave >> 1) * 16;        // sub-tile row offset
    const int wc   = (wave & 1) * 16;         // sub-tile col offset
    const int hsel = lane >> 4;               // half-wave select
    const int m    = lane & 15;

    v8f accM = {};
    v8f accA = {};

    #pragma unroll
    for (int k0 = 0; k0 < D; k0 += 32) {
        FragU a1, a2, b;
        const int ca0 = k0 + hsel * 8;        // A frag: K in [ca0, ca0+8)
        const int ca1 = k0 + 16 + hsel * 8;   //         K in [ca1, ca1+8)
        const int cb  = k0 + hsel * 16;       // B frag: same K as A frag
        a1.q[0] = *(const uint4*)&sGr[wr + m][ca0];
        a1.q[1] = *(const uint4*)&sGr[wr + m][ca1];
        a2.q[0] = *(const uint4*)&sSc[wr + m][ca0];
        a2.q[1] = *(const uint4*)&sSc[wr + m][ca1];
        b.q[0]  = *(const uint4*)&sGc[wc + m][ca0];
        b.q[1]  = *(const uint4*)&sGc[wc + m][ca1];

        accM = __builtin_amdgcn_wmma_f32_16x16x32_f16(
            false, a1.v, false, b.v, (short)0, accM, false, false);
        accA = __builtin_amdgcn_wmma_f32_16x16x32_f16(
            false, a2.v, false, b.v, (short)0, accA, false, false);
    }

    // Epilogue: C/D layout -> VGPR p holds row M = hsel*8 + p, col N = m
    const int   c_loc = wc + m;
    const int   cg    = c0 + c_loc;
    const float kc    = sKc[c_loc];
    const float t2c   = sT2c[c_loc];
    const float g2    = sG2;
    const size_t baseI = (size_t)i * N * N;

    #pragma unroll
    for (int p = 0; p < 8; ++p) {
        const int   r_loc = wr + hsel * 8 + p;
        const int   rg    = r0 + r_loc;
        const float kr    = sKr[r_loc];
        const float t2r   = sT2r[r_loc];
        const float Cr    = sCr[r_loc];
        const float Mv    = accM[p];
        const float Av    = accA[p];
        const size_t idx  = baseI + (size_t)rg * N + cg;
        __builtin_nontemporal_store(
            2.0f * (Cr + 4.0f * kr * Av * Mv) * kc, &out5[idx]);
        __builtin_nontemporal_store(
            4.0f * (4.0f * Mv * Mv - 2.0f * (t2r + t2c) + g2) * kr * kc, &out6[idx]);
    }
}

// ---------------------------------------------------------------------------
// Launch
// ---------------------------------------------------------------------------
extern "C" void kernel_launch(void* const* d_in, const int* in_sizes, int n_in,
                              void* d_out, int out_size, void* d_ws, size_t ws_size,
                              hipStream_t stream)
{
    (void)in_sizes; (void)n_in; (void)out_size; (void)ws_size;

    const float* x     = (const float*)d_in[0];
    const float* sigma = (const float*)d_in[1];
    const float* score = (const float*)d_in[2];
    const float* v     = (const float*)d_in[3];
    const float* gvs   = (const float*)d_in[4];

    float* out = (float*)d_out;
    constexpr size_t NN = (size_t)N * N;
    float* o_k   = out;                  // [N,N]
    float* o_kxj = o_k   + NN;           // [N,N,D]
    float* o_kxi = o_kxj + NN * D;       // [N,N,D]
    float* o_tr  = o_kxi + NN * D;       // [N,N]
    float* o_s5  = o_tr  + NN;           // [N,N,N]
    float* o_s6  = o_s5  + NN * N;       // [N,N,N]
    float* o_v   = o_s6  + NN * N;       // [N,N]

    float* ws    = (float*)d_ws;
    float* ws_t2 = ws;                   // [N,N]
    float* ws_C  = ws + NN;              // [N,N]

    // Kernel A: 8 waves/block, one wave per (i,j)
    pairwise_kernel<<<dim3(N / 8, N), 256, 0, stream>>>(
        x, sigma, score, v, gvs,
        o_k, o_kxj, o_kxi, o_tr, o_v, ws_t2, ws_C);

    // Kernel B: batch i in z, 32x32 (r,c) tiles, 4 waves/block
    nnn_kernel<<<dim3(N / 32, N / 32, N), 128, 0, stream>>>(
        x, sigma, score, o_k, ws_t2, ws_C, o_s5, o_s6);
}
